// SoftVM_62380105007442
// MI455X (gfx1250) — compile-verified, matrix operand order checked by reading, not run on hardware
//
#include <hip/hip_runtime.h>
#include <cstdint>

typedef __attribute__((ext_vector_type(16))) _Float16 v16h;
typedef __attribute__((ext_vector_type(8)))  float    v8f;
typedef __attribute__((ext_vector_type(4)))  float    f4;
typedef __attribute__((ext_vector_type(4)))  uint32_t u4;

#define NVALS           8192
#define ROWS_PER_INPUT  (NVALS * 4)            // 32768 one-hot byte rows per input
#define WAVES_PER_INPUT (ROWS_PER_INPUT / 16)  // 2048 waves (16 rows per wave tile)

// ---------------------------------------------------------------------------
// Kernel 1: WMMA bit-extraction decode.
//   C[m, j] = sum_k X[m, k] * ((k >> j) & 1)   (j = 0..7; columns 8..15 unused)
// One wave handles a 16-row tile, K=256 via 8 chained v_wmma_f32_16x16x32_f16.
//
// B-operand hoisting: with k = k0 | kl (k0 = step*32, kl < 32),
//   j < 5      : bit depends only on kl  -> step-invariant (in Bbase)
//   j = 5..7   : bit = (step >> (j-5))&1 -> same for all 16 halves of the lane
//   j >= 8     : 0
// so per step Bop = Bbase + splat(hb), hb = hsel * ((step >> (j-5)) & 1).
// ---------------------------------------------------------------------------
__global__ __launch_bounds__(256) void softvm_decode_wmma(
    const float* __restrict__ a, const float* __restrict__ b,
    uint32_t* __restrict__ aval, uint32_t* __restrict__ bval)
{
    const int gtid = blockIdx.x * 256 + threadIdx.x;
    const int wave = gtid >> 5;
    const int lane = threadIdx.x & 31;

    const float* src = a;
    uint32_t*    dst = aval;
    int          lw  = wave;
    if (wave >= WAVES_PER_INPUT) { src = b; dst = bval; lw = wave - WAVES_PER_INPUT; }

    const int row_base = lw * 16;                  // 16 one-hot byte rows, 256 f32 each
    const int m        = lane & 15;                // A-matrix row handled by this lane
    const int hi8      = (lane >> 4) * 8;          // K sub-block select (ISA A/B layout)
    const int j        = lane & 15;                // B-matrix column = bit index
    const float* rowp  = src + (size_t)(row_base + m) * 256;

    __builtin_prefetch(rowp, 0, 1);                // global_prefetch_b8 (speculative)

    // Step-invariant part of the B operand (j<5 bit pattern; 0 in cols 5..15).
    v16h Bbase;
#pragma unroll
    for (int h = 0; h < 8; ++h) {
        const int kl  = hi8 + h;                   // halves 0..7
        const int kl2 = 16 + hi8 + h;              // halves 8..15
        Bbase[h]     = (_Float16)(float)((j < 8) ? ((kl  >> j) & 1) : 0);
        Bbase[h + 8] = (_Float16)(float)((j < 8) ? ((kl2 >> j) & 1) : 0);
    }
    const float hsel = (j >= 5 && j < 8) ? 1.0f : 0.0f;  // high-bit column mask
    const int   sh   = (j - 5) & 31;

    v8f c = {0.f, 0.f, 0.f, 0.f, 0.f, 0.f, 0.f, 0.f};

#pragma unroll
    for (int step = 0; step < 8; ++step) {
        const int k0 = step * 32;
        // A tile: 16-bit A layout -> lane 0-15: halves = K k0+0..7 / k0+16..23;
        //                            lane 16-31: halves = K k0+8..15 / k0+24..31.
        f4 g0a = __builtin_nontemporal_load((const f4*)(rowp + k0 + hi8));
        f4 g0b = __builtin_nontemporal_load((const f4*)(rowp + k0 + hi8 + 4));
        f4 g1a = __builtin_nontemporal_load((const f4*)(rowp + k0 + 16 + hi8));
        f4 g1b = __builtin_nontemporal_load((const f4*)(rowp + k0 + 16 + hi8 + 4));

        v16h Aop;
        Aop[0]  = (_Float16)g0a.x;  Aop[1]  = (_Float16)g0a.y;
        Aop[2]  = (_Float16)g0a.z;  Aop[3]  = (_Float16)g0a.w;
        Aop[4]  = (_Float16)g0b.x;  Aop[5]  = (_Float16)g0b.y;
        Aop[6]  = (_Float16)g0b.z;  Aop[7]  = (_Float16)g0b.w;
        Aop[8]  = (_Float16)g1a.x;  Aop[9]  = (_Float16)g1a.y;
        Aop[10] = (_Float16)g1a.z;  Aop[11] = (_Float16)g1a.w;
        Aop[12] = (_Float16)g1b.x;  Aop[13] = (_Float16)g1b.y;
        Aop[14] = (_Float16)g1b.z;  Aop[15] = (_Float16)g1b.w;

        // Per-step high-bit broadcast (Bbase is 0 in columns 5..7, so add == select).
        const _Float16 hb = (_Float16)(hsel * (float)((step >> sh) & 1));
        const v16h Bop = Bbase + hb;               // 8x v_pk_add_f16

        c = __builtin_amdgcn_wmma_f32_16x16x32_f16(
                /*neg_a=*/false, Aop, /*neg_b=*/false, Bop,
                /*c_mod=*/(short)0, c, /*reuse_a=*/false, /*reuse_b=*/false);
    }

    // C layout: VGPR r, lanes 0-15 -> M=r; lanes 16-31 -> M=8+r. Column = bit j.
    // ballot bit j  -> bit j of row r's byte; bit 16+j -> bit j of row 8+r's byte.
    uint32_t by[16];
#pragma unroll
    for (int r = 0; r < 8; ++r) {
        const uint32_t msk = __builtin_amdgcn_ballot_w32(c[r] > 0.5f);
        by[r]     = msk & 0xFFu;
        by[r + 8] = (msk >> 16) & 0xFFu;
    }

    // 16 rows == 4 uint32 values (byte 0 = LSB, matching reference encode()).
    if (lane == 0) {
        u4 outv;
        outv.x = by[0]  | (by[1]  << 8) | (by[2]  << 16) | (by[3]  << 24);
        outv.y = by[4]  | (by[5]  << 8) | (by[6]  << 16) | (by[7]  << 24);
        outv.z = by[8]  | (by[9]  << 8) | (by[10] << 16) | (by[11] << 24);
        outv.w = by[12] | (by[13] << 8) | (by[14] << 16) | (by[15] << 24);
        *(u4*)(dst + lw * 4) = outv;
    }
}

// ---------------------------------------------------------------------------
// Kernel 2: integer ALU + one-hot streaming emit (bandwidth bound, 235 MB).
// One block per value; each thread writes one float4 per op (NT b128 stores).
// ---------------------------------------------------------------------------
__global__ __launch_bounds__(256) void softvm_emit(
    const uint32_t* __restrict__ aval, const uint32_t* __restrict__ bval,
    float* __restrict__ out)
{
    const int v = blockIdx.x;
    const uint32_t A  = aval[v];
    const uint32_t Bv = bval[v];

    uint32_t r[7];
    r[0] = A + Bv;                        // soft_add
    r[1] = A - Bv;                        // soft_add(a, soft_negate(b)) == a-b mod 2^32
    r[2] = A * Bv;                        // uint32 wraparound multiply
    r[3] = (Bv == 0u) ? 0u : (A / Bv);    // guarded divide
    r[4] = A & Bv;
    r[5] = A | Bv;
    r[6] = A ^ Bv;

    const int byte_i = threadIdx.x >> 6;          // 0..3
    const int cbase  = (threadIdx.x & 63) * 4;    // float4 column base 0..252
    const size_t opStride = (size_t)NVALS * 4 * 256;
    float* base = out + ((size_t)v * 4 + byte_i) * 256 + cbase;

#pragma unroll
    for (int op = 0; op < 7; ++op) {
        const uint32_t hot = (r[op] >> (byte_i * 8)) & 255u;
        f4 val;
        val.x = (hot == (uint32_t)(cbase + 0)) ? 1.0f : 0.0f;
        val.y = (hot == (uint32_t)(cbase + 1)) ? 1.0f : 0.0f;
        val.z = (hot == (uint32_t)(cbase + 2)) ? 1.0f : 0.0f;
        val.w = (hot == (uint32_t)(cbase + 3)) ? 1.0f : 0.0f;
        __builtin_nontemporal_store(val, (f4*)(base + op * opStride));
    }
}

// ---------------------------------------------------------------------------
extern "C" void kernel_launch(void* const* d_in, const int* in_sizes, int n_in,
                              void* d_out, int out_size, void* d_ws, size_t ws_size,
                              hipStream_t stream)
{
    const float* a = (const float*)d_in[0];
    const float* b = (const float*)d_in[1];

    uint32_t* aval = (uint32_t*)d_ws;        // 32 KB
    uint32_t* bval = aval + NVALS;           // 32 KB

    // 2 inputs * 2048 wave-tiles, 8 waves per 256-thread block -> 512 blocks.
    softvm_decode_wmma<<<512, 256, 0, stream>>>(a, b, aval, bval);

    // One block per value: 8192 blocks, each streams 28 KB of one-hot output.
    softvm_emit<<<NVALS, 256, 0, stream>>>(aval, bval, (float*)d_out);
}
